// FuseformerModel_39109972198040
// MI455X (gfx1250) — matrix operand order repaired and checked by reading, not compiled, not run on hardware
//
#include <hip/hip_runtime.h>
#include <hip/hip_bf16.h>
#include <math.h>

// ---------------- model constants ----------------
#define D_MODEL 768
#define HE_     12
#define HH_     4
#define NLZ     32      // lorentz spatial head dim
#define FF_     3072
#define GATE_   64
#define B_      2
#define TS_     256
#define TT_     256
#define V_      32000
#define NTOK    (B_ * TS_)   // 512 (enc and dec both 512)
#define DHE     64           // D/HE

typedef __attribute__((ext_vector_type(16))) _Float16 v16h;
typedef __attribute__((ext_vector_type(8)))  float    v8f;

union V16H { v16h v; _Float16 e[16]; };

static __device__ __forceinline__ int imin(int a, int b) { return a < b ? a : b; }

static __device__ __forceinline__ v16h cvt16(float4 x0, float4 x1, float4 x2, float4 x3)
{
    V16H r;
    r.e[0]  = (_Float16)x0.x; r.e[1]  = (_Float16)x0.y;
    r.e[2]  = (_Float16)x0.z; r.e[3]  = (_Float16)x0.w;
    r.e[4]  = (_Float16)x1.x; r.e[5]  = (_Float16)x1.y;
    r.e[6]  = (_Float16)x1.z; r.e[7]  = (_Float16)x1.w;
    r.e[8]  = (_Float16)x2.x; r.e[9]  = (_Float16)x2.y;
    r.e[10] = (_Float16)x2.z; r.e[11] = (_Float16)x2.w;
    r.e[12] = (_Float16)x3.x; r.e[13] = (_Float16)x3.y;
    r.e[14] = (_Float16)x3.z; r.e[15] = (_Float16)x3.w;
    return r.v;
}

// B fragment for row-major (K,N) weights: 16 strided loads, coalesced across lanes
static __device__ __forceinline__ v16h loadB_strided(const float* __restrict__ wp, int N)
{
    V16H b;
    #pragma unroll
    for (int h = 0; h < 16; ++h) b.e[h] = (_Float16)wp[(size_t)h * N];
    return b.v;
}

// B fragment for row-major (N,K) weights (transB): 16 contiguous floats
static __device__ __forceinline__ v16h loadB_contig(const float* __restrict__ wp)
{
    return cvt16(*reinterpret_cast<const float4*>(wp),
                 *reinterpret_cast<const float4*>(wp + 4),
                 *reinterpret_cast<const float4*>(wp + 8),
                 *reinterpret_cast<const float4*>(wp + 12));
}

static __device__ __forceinline__ float apply_act(float vv, int act)
{
    if (act == 1) {                 // gelu (tanh approx, jax default)
        float x3 = vv * vv * vv;
        vv = 0.5f * vv * (1.f + tanhf(0.7978845608028654f * (vv + 0.044715f * x3)));
    } else if (act == 2) {
        vv = tanhf(vv);
    } else if (act == 3) {
        vv = 1.f / (1.f + __expf(-vv));
    }
    return vv;
}

// ======================================================================
// Interior WMMA GEMM: requires M%16==0 and N%64==0 (K may be ragged).
// One wave computes a 16x64 slab of C: 4 unconditional v_wmma per K-step
// on four independent accumulators; A as 4x b128 loads; fp32->f16 cvt.
// C = act(A[M,K] @ W + bias) (+= C if accum); W (K,N), or (N,K) if TRANSB.
// ======================================================================
template<int TRANSB>
__global__ __launch_bounds__(32)
void gemm_wmma_fast(const float* __restrict__ A, const float* __restrict__ W,
                    const float* __restrict__ bias, float* __restrict__ C,
                    int M, int N, int K, int ldA, int ldC, int act, int accum)
{
    const int lane  = threadIdx.x & 31;
    const int row   = lane & 15;
    const int grp   = lane >> 4;          // lane half: 0 or 1
    const int m0    = blockIdx.y * 16;
    const int nbase = blockIdx.x * 64;
    const int am    = m0 + row;           // A row fed by this lane

    // A element map: half h -> K = k0 + h + 16*grp
    const float* ap = A + (size_t)am * ldA + (grp << 4);
    // B element map: half h -> K = k0 + h + 16*grp ; column = n0 + row
    const float* wb0 = TRANSB ? (W + (size_t)(nbase + row) * K + (grp << 4))
                              : (W + (size_t)(grp << 4) * N + nbase + row);

    v8f acc0 = {}, acc1 = {}, acc2 = {}, acc3 = {};

    const int kfull = K & ~31;
    for (int k0 = 0; k0 < kfull; k0 += 32) {
        const v16h a = cvt16(*reinterpret_cast<const float4*>(ap),
                             *reinterpret_cast<const float4*>(ap + 4),
                             *reinterpret_cast<const float4*>(ap + 8),
                             *reinterpret_cast<const float4*>(ap + 12));
        ap += 32;
        v16h b0, b1, b2, b3;
        if (TRANSB) {
            const float* wp = wb0 + k0;
            b0 = loadB_contig(wp);
            b1 = loadB_contig(wp + (size_t)16 * K);
            b2 = loadB_contig(wp + (size_t)32 * K);
            b3 = loadB_contig(wp + (size_t)48 * K);
        } else {
            const float* wp = wb0 + (size_t)k0 * N;
            b0 = loadB_strided(wp,      N);
            b1 = loadB_strided(wp + 16, N);
            b2 = loadB_strided(wp + 32, N);
            b3 = loadB_strided(wp + 48, N);
        }
        acc0 = __builtin_amdgcn_wmma_f32_16x16x32_f16(false, a, false, b0, (short)0, acc0, false, false);
        acc1 = __builtin_amdgcn_wmma_f32_16x16x32_f16(false, a, false, b1, (short)0, acc1, false, false);
        acc2 = __builtin_amdgcn_wmma_f32_16x16x32_f16(false, a, false, b2, (short)0, acc2, false, false);
        acc3 = __builtin_amdgcn_wmma_f32_16x16x32_f16(false, a, false, b3, (short)0, acc3, false, false);
    }
    if (kfull < K) {                       // ragged-K tail (e.g. K=132): clamp+select
        const int k0 = kfull;
        V16H a;
        #pragma unroll
        for (int h = 0; h < 16; ++h) {
            int ka = k0 + h + (grp << 4);
            float av = A[(size_t)am * ldA + imin(ka, K - 1)];
            a.e[h] = (_Float16)((ka < K) ? av : 0.f);
        }
        V16H bt[4];
        #pragma unroll
        for (int nt = 0; nt < 4; ++nt) {
            const int bn = nbase + nt * 16 + row;
            #pragma unroll
            for (int h = 0; h < 16; ++h) {
                int kb  = k0 + h + (grp << 4);
                int kbC = imin(kb, K - 1);
                float wv = TRANSB ? W[(size_t)bn * K + kbC] : W[(size_t)kbC * N + bn];
                bt[nt].e[h] = (_Float16)((kb < K) ? wv : 0.f);
            }
        }
        acc0 = __builtin_amdgcn_wmma_f32_16x16x32_f16(false, a.v, false, bt[0].v, (short)0, acc0, false, false);
        acc1 = __builtin_amdgcn_wmma_f32_16x16x32_f16(false, a.v, false, bt[1].v, (short)0, acc1, false, false);
        acc2 = __builtin_amdgcn_wmma_f32_16x16x32_f16(false, a.v, false, bt[2].v, (short)0, acc2, false, false);
        acc3 = __builtin_amdgcn_wmma_f32_16x16x32_f16(false, a.v, false, bt[3].v, (short)0, acc3, false, false);
    }

    // epilogue: C/D layout VGPR r -> M = m0 + r + 8*grp, N = n0 + (lane&15)
    v8f accs[4] = {acc0, acc1, acc2, acc3};
    #pragma unroll
    for (int nt = 0; nt < 4; ++nt) {
        const int nn = nbase + nt * 16 + row;
        const float bv = bias ? bias[nn] : 0.f;
        float* cp = C + (size_t)(m0 + (grp << 3)) * ldC + nn;
        #pragma unroll
        for (int r = 0; r < 8; ++r) {
            float vv = apply_act(accs[nt][r] + bv, act);
            if (accum) vv += cp[(size_t)r * ldC];
            cp[(size_t)r * ldC] = vv;
        }
    }
}

// ======================================================================
// Generic guarded WMMA GEMM (16x16 per wave, clamp+select): used only
// for ragged-N cases (fgo: N=1).
// ======================================================================
__global__ __launch_bounds__(32)
void gemm_wmma_generic(const float* __restrict__ A, const float* __restrict__ W,
                       const float* __restrict__ bias, float* __restrict__ C,
                       int M, int N, int K, int ldA, int ldC,
                       int transB, int act, int accum)
{
    const int lane = threadIdx.x & 31;
    const int row  = lane & 15;
    const int grp  = lane >> 4;
    const int m0   = blockIdx.y * 16;
    const int n0   = blockIdx.x * 16;
    const int am   = m0 + row;
    const int amC  = imin(am, M - 1);
    const bool amok = (am < M);
    const int bn   = n0 + row;
    const int bnC  = imin(bn, N - 1);
    const bool bnok = (bn < N);

    v8f acc = {};
    for (int k0 = 0; k0 < K; k0 += 32) {
        V16H a, b;
        #pragma unroll
        for (int h = 0; h < 16; ++h) {
            int ka = k0 + h + (grp << 4);
            float av = A[(size_t)amC * ldA + imin(ka, K - 1)];
            a.e[h] = (_Float16)((amok && ka < K) ? av : 0.f);
            int kb  = k0 + h + (grp << 4);
            int kbC = imin(kb, K - 1);
            float wv = transB ? W[(size_t)bnC * K + kbC] : W[(size_t)kbC * N + bnC];
            b.e[h] = (_Float16)((bnok && kb < K) ? wv : 0.f);
        }
        acc = __builtin_amdgcn_wmma_f32_16x16x32_f16(false, a.v, false, b.v,
                                                     (short)0, acc, false, false);
    }
    const float bv = (bias && bnok) ? bias[bnC] : 0.f;
    #pragma unroll
    for (int r = 0; r < 8; ++r) {
        const int mm = m0 + r + (grp << 3);
        if (mm < M && bnok) {
            float vv = apply_act(acc[r] + bv, act);
            float* cp = C + (size_t)mm * ldC + bn;
            if (accum) vv += *cp;
            *cp = vv;
        }
    }
}

// ======================================================================
// Embedding gather: one block per token
// ======================================================================
__global__ __launch_bounds__(256)
void embed_kernel(const float* __restrict__ E, const int* __restrict__ ids,
                  float* __restrict__ out)
{
    int t  = blockIdx.x;
    int id = ids[t];
    const float* src = E + (size_t)id * D_MODEL;
    float* dst = out + (size_t)t * D_MODEL;
    for (int d = threadIdx.x; d < D_MODEL; d += 256) dst[d] = src[d];
}

// ======================================================================
// LayerNorm: one block (256 thr) per token row
// ======================================================================
__global__ __launch_bounds__(256)
void layernorm_kernel(const float* __restrict__ x, const float* __restrict__ g,
                      const float* __restrict__ b, float* __restrict__ y)
{
    __shared__ float red[256];
    int t = blockIdx.x;
    int tid = threadIdx.x;
    const float* xr = x + (size_t)t * D_MODEL;
    float s = 0.f;
    for (int d = tid; d < D_MODEL; d += 256) s += xr[d];
    red[tid] = s; __syncthreads();
    for (int st = 128; st > 0; st >>= 1) { if (tid < st) red[tid] += red[tid + st]; __syncthreads(); }
    float mean = red[0] / (float)D_MODEL;
    __syncthreads();
    float vs = 0.f;
    for (int d = tid; d < D_MODEL; d += 256) { float u = xr[d] - mean; vs += u * u; }
    red[tid] = vs; __syncthreads();
    for (int st = 128; st > 0; st >>= 1) { if (tid < st) red[tid] += red[tid + st]; __syncthreads(); }
    float rstd = rsqrtf(red[0] / (float)D_MODEL + 1e-5f);
    for (int d = tid; d < D_MODEL; d += 256)
        y[(size_t)t * D_MODEL + d] = (xr[d] - mean) * rstd * g[d] + b[d];
}

// ======================================================================
// Euclidean softmax attention: one block per (b, head, query-row)
// q: (B*Tq, 768)  k,v: (B*Tk, 768)  o: (B*Tq, 768), head dim 64
// ======================================================================
__global__ __launch_bounds__(256)
void euclid_attn_kernel(const float* __restrict__ q, const float* __restrict__ k,
                        const float* __restrict__ v, float* __restrict__ o,
                        int Tq, int Tk, int causal)
{
    int i  = blockIdx.x % Tq;
    int bh = blockIdx.x / Tq;
    int h  = bh % HE_;
    int b  = bh / HE_;
    __shared__ float a[256];
    __shared__ float red[256];
    __shared__ float qrow[DHE];
    int tid = threadIdx.x;
    const float* qp = q + ((size_t)(b * Tq + i)) * D_MODEL + h * DHE;
    if (tid < DHE) qrow[tid] = qp[tid];
    __syncthreads();
    float s = -1e30f;
    if (tid < Tk && !(causal && tid > i)) {
        const float* kp = k + ((size_t)(b * Tk + tid)) * D_MODEL + h * DHE;
        float d0 = 0.f;
        #pragma unroll 8
        for (int d = 0; d < DHE; ++d) d0 += qrow[d] * kp[d];
        s = d0 * 0.125f;                       // 1/sqrt(64)
    }
    red[tid] = s; __syncthreads();
    for (int st = 128; st > 0; st >>= 1) { if (tid < st) red[tid] = fmaxf(red[tid], red[tid + st]); __syncthreads(); }
    float mx = red[0]; __syncthreads();
    float e = (tid < Tk) ? __expf(s - mx) : 0.f;
    red[tid] = e; __syncthreads();
    for (int st = 128; st > 0; st >>= 1) { if (tid < st) red[tid] += red[tid + st]; __syncthreads(); }
    a[tid] = e / red[0];
    __syncthreads();
    // weighted sum over values: 4 partial chunks of Tk/4 each
    int d    = tid & (DHE - 1);
    int part = tid >> 6;
    int span = Tk >> 2;
    float accv = 0.f;
    for (int j = part * span; j < (part + 1) * span && j < Tk; ++j)
        accv += a[j] * v[((size_t)(b * Tk + j)) * D_MODEL + h * DHE + d];
    red[tid] = accv; __syncthreads();
    if (tid < DHE)
        o[((size_t)(b * Tq + i)) * D_MODEL + h * DHE + tid] =
            red[tid] + red[tid + 64] + red[tid + 128] + red[tid + 192];
}

// ======================================================================
// Lorentz (hyperboloid) attention, KSTEPS=1, TAU=1.
// uq: (B*Tq, 128)  uk,uv: (B*Tk, 128)  ylog: (B*Tq, 132)
// One block per (b, head, query-row).
// ======================================================================
__global__ __launch_bounds__(256)
void lorentz_attn_kernel(const float* __restrict__ uq, const float* __restrict__ uk,
                         const float* __restrict__ uv, float* __restrict__ ylog,
                         int Tq, int Tk, int causal)
{
    const float MINC = 1.f + 1e-6f;
    int i  = blockIdx.x % Tq;
    int bh = blockIdx.x / Tq;
    int h  = bh % HH_;
    int b  = bh / HH_;
    const int Dm = HH_ * NLZ;     // 128
    __shared__ float qh[NLZ + 1];
    __shared__ float w[256];
    __shared__ float tv[256];
    __shared__ float red[256];
    __shared__ float stepv[NLZ + 1];
    __shared__ float muv[NLZ + 1];
    __shared__ float scal[1];
    int tid = threadIdx.x;
    // lift query to hyperboloid (time-first)
    if (tid == 0) {
        const float* qp = uq + ((size_t)(b * Tq + i)) * Dm + h * NLZ;
        float ss = 0.f;
        for (int c = 0; c < NLZ; ++c) { float u = qp[c]; qh[c + 1] = u; ss += u * u; }
        qh[0] = sqrtf(1.f + ss);
    }
    __syncthreads();
    // scores from geodesic distance, plus value time components
    float s = -1e30f;
    if (tid < Tk) {
        const float* kp = uk + ((size_t)(b * Tk + tid)) * Dm + h * NLZ;
        const float* vp = uv + ((size_t)(b * Tk + tid)) * Dm + h * NLZ;
        float ssk = 0.f, ipk = 0.f, ssv = 0.f;
        for (int c = 0; c < NLZ; ++c) {
            float kv = kp[c]; ssk += kv * kv; ipk += qh[c + 1] * kv;
            float vv = vp[c]; ssv += vv * vv;
        }
        tv[tid] = sqrtf(1.f + ssv);
        float ip = ipk - qh[0] * sqrtf(1.f + ssk);
        float al = fmaxf(-ip, MINC);
        float dist = acoshf(al);
        s = -(dist * dist);                  // TAU = 1
        if (causal && tid > i) s = -1e30f;
    }
    red[tid] = s; __syncthreads();
    for (int st = 128; st > 0; st >>= 1) { if (tid < st) red[tid] = fmaxf(red[tid], red[tid + st]); __syncthreads(); }
    float mx = red[0]; __syncthreads();
    float e = (tid < Tk) ? __expf(s - mx) : 0.f;
    red[tid] = e; __syncthreads();
    for (int st = 128; st > 0; st >>= 1) { if (tid < st) red[tid] += red[tid + st]; __syncthreads(); }
    float aj = e / red[0];
    __syncthreads();
    // logmap weights: w_j = a_j * d/sqrt(alpha^2-1), sum of w_j*alpha_j
    float wj = 0.f, waj = 0.f;
    if (tid < Tk) {
        const float* vp = uv + ((size_t)(b * Tk + tid)) * Dm + h * NLZ;
        float ipv = 0.f;
        for (int c = 0; c < NLZ; ++c) ipv += qh[c + 1] * vp[c];
        float mk = ipv - qh[0] * tv[tid];     // <mu, vh_j>_L
        float al = fmaxf(-mk, MINC);
        float dd = acoshf(al);
        float coef = dd * rsqrtf(fmaxf(al * al - 1.f, 1e-12f));
        wj  = aj * coef;
        waj = wj * al;
    }
    w[tid]   = wj;
    red[tid] = waj; __syncthreads();
    for (int st = 128; st > 0; st >>= 1) { if (tid < st) red[tid] += red[tid + st]; __syncthreads(); }
    float wa_sum = red[0];
    __syncthreads();
    // step[c] = sum_j w_j * vh_j[c] - wa_sum * mu[c]
    if (tid <= NLZ) {
        float acc2 = 0.f;
        if (tid == 0) {
            for (int j = 0; j < Tk; ++j) acc2 += w[j] * tv[j];
        } else {
            const float* vb = uv + ((size_t)(b * Tk)) * Dm + h * NLZ + (tid - 1);
            for (int j = 0; j < Tk; ++j) acc2 += w[j] * vb[(size_t)j * Dm];
        }
        stepv[tid] = acc2 - wa_sum * qh[tid];
    }
    __syncthreads();
    // expmap at mu=qh, then logmap at origin
    if (tid == 0) {
        float mk = -stepv[0] * stepv[0];
        for (int c = 1; c <= NLZ; ++c) mk += stepv[c] * stepv[c];
        float vn = sqrtf(fmaxf(mk, 1e-12f));
        float ch = coshf(vn), sh = sinhf(vn) / vn;
        for (int c = 0; c <= NLZ; ++c) muv[c] = ch * qh[c] + sh * stepv[c];
        float al = fmaxf(muv[0], MINC);
        scal[0] = acoshf(al) * rsqrtf(fmaxf(al * al - 1.f, 1e-12f));
    }
    __syncthreads();
    if (tid <= NLZ) {
        float yv = (tid == 0) ? 0.f : scal[0] * muv[tid];
        ylog[((size_t)(b * Tq + i)) * (HH_ * (NLZ + 1)) + h * (NLZ + 1) + tid] = yv;
    }
}

// ======================================================================
// Gated fusion residual: x += a*h2 + (1-a)*e2, cat = [e2 | h2] (T,1536)
// ======================================================================
__global__ __launch_bounds__(256)
void fusion_add_kernel(float* __restrict__ x, const float* __restrict__ cat,
                       const float* __restrict__ alpha)
{
    int t = blockIdx.x;
    float a = alpha[t];
    const float* e2 = cat + (size_t)t * (2 * D_MODEL);
    const float* h2 = e2 + D_MODEL;
    float* xr = x + (size_t)t * D_MODEL;
    for (int d = threadIdx.x; d < D_MODEL; d += 256)
        xr[d] += a * h2[d] + (1.f - a) * e2[d];
}

// ======================================================================
// Host side
// ======================================================================
struct Dp { const float *b, *w; };
struct Lp { const float *b, *g; };

struct DecL {
    Dp cek, ceo, ceq, cev, chk, cho, chq, chv;
    Dp fcUe, fcUh, fcgh, fcgo, ff1, ff2, fsUe, fsUh, fsgh, fsgo;
    Lp ln_cross, ln_ffn, ln_self;
    Dp sek, seo, seq, sev, shk, sho, shq, shv;
};
struct EncL {
    Dp ek, eo, eq, ev, fUe, fUh, ff1, ff2, fgh, fgo, hk, ho, hq, hv;
    Lp ln1, ln2;
};

static inline void gemm(hipStream_t st, const float* A, const float* W, const float* bias,
                        float* C, int M, int N, int K, int ldA, int ldC,
                        int transB, int act, int accum)
{
    if ((M % 16 == 0) && (N % 64 == 0)) {
        dim3 g(N / 64, M / 16);
        if (transB)
            gemm_wmma_fast<1><<<g, dim3(32), 0, st>>>(A, W, bias, C, M, N, K, ldA, ldC, act, accum);
        else
            gemm_wmma_fast<0><<<g, dim3(32), 0, st>>>(A, W, bias, C, M, N, K, ldA, ldC, act, accum);
    } else {
        dim3 g((N + 15) / 16, (M + 15) / 16);
        gemm_wmma_generic<<<g, dim3(32), 0, st>>>(A, W, bias, C, M, N, K, ldA, ldC, transB, act, accum);
    }
}

extern "C" void kernel_launch(void* const* d_in, const int* in_sizes, int n_in,
                              void* d_out, int out_size, void* d_ws, size_t ws_size,
                              hipStream_t stream)
{
    (void)in_sizes; (void)n_in; (void)out_size; (void)ws_size;
    const int* input_ids = (const int*)d_in[0];
    const int* dec_ids   = (const int*)d_in[1];

    int idx = 2;
    auto DP = [&](void) { Dp p; p.b = (const float*)d_in[idx++]; p.w = (const float*)d_in[idx++]; return p; };
    auto LP = [&](void) { Lp p; p.b = (const float*)d_in[idx++]; p.g = (const float*)d_in[idx++]; return p; };

    // pytree order: params dict sorted keys: "dec" < "embed" < "enc";
    // per-layer dict keys sorted (ASCII, uppercase before lowercase).
    DecL dec[2];
    for (int l = 0; l < 2; ++l) {
        DecL& L = dec[l];
        L.cek = DP(); L.ceo = DP(); L.ceq = DP(); L.cev = DP();
        L.chk = DP(); L.cho = DP(); L.chq = DP(); L.chv = DP();
        L.fcUe = DP(); L.fcUh = DP(); L.fcgh = DP(); L.fcgo = DP();
        L.ff1 = DP(); L.ff2 = DP();
        L.fsUe = DP(); L.fsUh = DP(); L.fsgh = DP(); L.fsgo = DP();
        L.ln_cross = LP(); L.ln_ffn = LP(); L.ln_self = LP();
        L.sek = DP(); L.seo = DP(); L.seq = DP(); L.sev = DP();
        L.shk = DP(); L.sho = DP(); L.shq = DP(); L.shv = DP();
    }
    const float* embed = (const float*)d_in[idx++];
    EncL enc[2];
    for (int l = 0; l < 2; ++l) {
        EncL& L = enc[l];
        L.ek = DP(); L.eo = DP(); L.eq = DP(); L.ev = DP();
        L.fUe = DP(); L.fUh = DP(); L.ff1 = DP(); L.ff2 = DP();
        L.fgh = DP(); L.fgo = DP();
        L.hk = DP(); L.ho = DP(); L.hq = DP(); L.hv = DP();
        L.ln1 = LP(); L.ln2 = LP();
    }

    // ----- workspace carve-up (floats) -----
    float* ws = (float*)d_ws;
    size_t off = 0;
    auto AL = [&](size_t n) { float* p = ws + off; off += n; return p; };
    float* xe   = AL((size_t)NTOK * D_MODEL);   // encoder stream
    float* xd   = AL((size_t)NTOK * D_MODEL);   // decoder stream
    float* hbuf = AL((size_t)NTOK * D_MODEL);   // LN output
    float* qb   = AL((size_t)NTOK * D_MODEL);
    float* kb   = AL((size_t)NTOK * D_MODEL);
    float* vb   = AL((size_t)NTOK * D_MODEL);
    float* att  = AL((size_t)NTOK * D_MODEL);
    float* eout = AL((size_t)NTOK * D_MODEL);
    float* uqb  = AL((size_t)NTOK * (HH_ * NLZ));
    float* ukb  = AL((size_t)NTOK * (HH_ * NLZ));
    float* uvb  = AL((size_t)NTOK * (HH_ * NLZ));
    float* ylg  = AL((size_t)NTOK * (HH_ * (NLZ + 1)));
    float* hout = AL((size_t)NTOK * D_MODEL);
    float* catb = AL((size_t)NTOK * (2 * D_MODEL));
    float* gb   = AL((size_t)NTOK * GATE_);
    float* alb  = AL((size_t)NTOK);
    float* ffb  = AL((size_t)NTOK * FF_);

    const int M = NTOK;                 // 512 rows for both streams
    const int Dh = D_MODEL;
    const int HN = HH_ * NLZ;           // 128
    const int HN1 = HH_ * (NLZ + 1);    // 132

    // ---------------- embeddings ----------------
    embed_kernel<<<NTOK, 256, 0, stream>>>(embed, input_ids, xe);
    embed_kernel<<<NTOK, 256, 0, stream>>>(embed, dec_ids,   xd);

    // ---------------- encoder ----------------
    for (int l = 0; l < 2; ++l) {
        EncL& L = enc[l];
        layernorm_kernel<<<NTOK, 256, 0, stream>>>(xe, L.ln1.g, L.ln1.b, hbuf);
        gemm(stream, hbuf, L.eq.w, L.eq.b, qb, M, Dh, Dh, Dh, Dh, 0, 0, 0);
        gemm(stream, hbuf, L.ek.w, L.ek.b, kb, M, Dh, Dh, Dh, Dh, 0, 0, 0);
        gemm(stream, hbuf, L.ev.w, L.ev.b, vb, M, Dh, Dh, Dh, Dh, 0, 0, 0);
        euclid_attn_kernel<<<B_ * HE_ * TS_, 256, 0, stream>>>(qb, kb, vb, att, TS_, TS_, 0);
        gemm(stream, att, L.eo.w, L.eo.b, eout, M, Dh, Dh, Dh, Dh, 0, 0, 0);
        gemm(stream, hbuf, L.hq.w, L.hq.b, uqb, M, HN, Dh, Dh, HN, 0, 0, 0);
        gemm(stream, hbuf, L.hk.w, L.hk.b, ukb, M, HN, Dh, Dh, HN, 0, 0, 0);
        gemm(stream, hbuf, L.hv.w, L.hv.b, uvb, M, HN, Dh, Dh, HN, 0, 0, 0);
        lorentz_attn_kernel<<<B_ * HH_ * TS_, 256, 0, stream>>>(uqb, ukb, uvb, ylg, TS_, TS_, 0);
        gemm(stream, ylg, L.ho.w, L.ho.b, hout, M, Dh, HN1, HN1, Dh, 0, 0, 0);
        gemm(stream, eout, L.fUe.w, L.fUe.b, catb,           M, Dh, Dh, Dh, 2 * Dh, 0, 0, 0);
        gemm(stream, hout, L.fUh.w, L.fUh.b, catb + D_MODEL, M, Dh, Dh, Dh, 2 * Dh, 0, 0, 0);
        gemm(stream, catb, L.fgh.w, L.fgh.b, gb, M, GATE_, 2 * Dh, 2 * Dh, GATE_, 0, 2, 0);
        gemm(stream, gb, L.fgo.w, L.fgo.b, alb, M, 1, GATE_, GATE_, 1, 0, 3, 0);
        fusion_add_kernel<<<NTOK, 256, 0, stream>>>(xe, catb, alb);
        layernorm_kernel<<<NTOK, 256, 0, stream>>>(xe, L.ln2.g, L.ln2.b, hbuf);
        gemm(stream, hbuf, L.ff1.w, L.ff1.b, ffb, M, FF_, Dh, Dh, FF_, 0, 1, 0);
        gemm(stream, ffb, L.ff2.w, L.ff2.b, xe, M, Dh, FF_, FF_, Dh, 0, 0, 1);
    }

    // ---------------- decoder ----------------
    for (int l = 0; l < 2; ++l) {
        DecL& L = dec[l];
        // masked self attention
        layernorm_kernel<<<NTOK, 256, 0, stream>>>(xd, L.ln_self.g, L.ln_self.b, hbuf);
        gemm(stream, hbuf, L.seq.w, L.seq.b, qb, M, Dh, Dh, Dh, Dh, 0, 0, 0);
        gemm(stream, hbuf, L.sek.w, L.sek.b, kb, M, Dh, Dh, Dh, Dh, 0, 0, 0);
        gemm(stream, hbuf, L.sev.w, L.sev.b, vb, M, Dh, Dh, Dh, Dh, 0, 0, 0);
        euclid_attn_kernel<<<B_ * HE_ * TT_, 256, 0, stream>>>(qb, kb, vb, att, TT_, TT_, 1);
        gemm(stream, att, L.seo.w, L.seo.b, eout, M, Dh, Dh, Dh, Dh, 0, 0, 0);
        gemm(stream, hbuf, L.shq.w, L.shq.b, uqb, M, HN, Dh, Dh, HN, 0, 0, 0);
        gemm(stream, hbuf, L.shk.w, L.shk.b, ukb, M, HN, Dh, Dh, HN, 0, 0, 0);
        gemm(stream, hbuf, L.shv.w, L.shv.b, uvb, M, HN, Dh, Dh, HN, 0, 0, 0);
        lorentz_attn_kernel<<<B_ * HH_ * TT_, 256, 0, stream>>>(uqb, ukb, uvb, ylg, TT_, TT_, 1);
        gemm(stream, ylg, L.sho.w, L.sho.b, hout, M, Dh, HN1, HN1, Dh, 0, 0, 0);
        gemm(stream, eout, L.fsUe.w, L.fsUe.b, catb,           M, Dh, Dh, Dh, 2 * Dh, 0, 0, 0);
        gemm(stream, hout, L.fsUh.w, L.fsUh.b, catb + D_MODEL, M, Dh, Dh, Dh, 2 * Dh, 0, 0, 0);
        gemm(stream, catb, L.fsgh.w, L.fsgh.b, gb, M, GATE_, 2 * Dh, 2 * Dh, GATE_, 0, 2, 0);
        gemm(stream, gb, L.fsgo.w, L.fsgo.b, alb, M, 1, GATE_, GATE_, 1, 0, 3, 0);
        fusion_add_kernel<<<NTOK, 256, 0, stream>>>(xd, catb, alb);
        // cross attention (kv from encoder output xe, not layernormed)
        layernorm_kernel<<<NTOK, 256, 0, stream>>>(xd, L.ln_cross.g, L.ln_cross.b, hbuf);
        gemm(stream, hbuf, L.ceq.w, L.ceq.b, qb, M, Dh, Dh, Dh, Dh, 0, 0, 0);
        gemm(stream, xe,   L.cek.w, L.cek.b, kb, M, Dh, Dh, Dh, Dh, 0, 0, 0);
        gemm(stream, xe,   L.cev.w, L.cev.b, vb, M, Dh, Dh, Dh, Dh, 0, 0, 0);
        euclid_attn_kernel<<<B_ * HE_ * TT_, 256, 0, stream>>>(qb, kb, vb, att, TT_, TS_, 0);
        gemm(stream, att, L.ceo.w, L.ceo.b, eout, M, Dh, Dh, Dh, Dh, 0, 0, 0);
        gemm(stream, hbuf, L.chq.w, L.chq.b, uqb, M, HN, Dh, Dh, HN, 0, 0, 0);
        gemm(stream, xe,   L.chk.w, L.chk.b, ukb, M, HN, Dh, Dh, HN, 0, 0, 0);
        gemm(stream, xe,   L.chv.w, L.chv.b, uvb, M, HN, Dh, Dh, HN, 0, 0, 0);
        lorentz_attn_kernel<<<B_ * HH_ * TT_, 256, 0, stream>>>(uqb, ukb, uvb, ylg, TT_, TS_, 0);
        gemm(stream, ylg, L.cho.w, L.cho.b, hout, M, Dh, HN1, HN1, Dh, 0, 0, 0);
        gemm(stream, eout, L.fcUe.w, L.fcUe.b, catb,           M, Dh, Dh, Dh, 2 * Dh, 0, 0, 0);
        gemm(stream, hout, L.fcUh.w, L.fcUh.b, catb + D_MODEL, M, Dh, Dh, Dh, 2 * Dh, 0, 0, 0);
        gemm(stream, catb, L.fcgh.w, L.fcgh.b, gb, M, GATE_, 2 * Dh, 2 * Dh, GATE_, 0, 2, 0);
        gemm(stream, gb, L.fcgo.w, L.fcgo.b, alb, M, 1, GATE_, GATE_, 1, 0, 3, 0);
        fusion_add_kernel<<<NTOK, 256, 0, stream>>>(xd, catb, alb);
        // FFN
        layernorm_kernel<<<NTOK, 256, 0, stream>>>(xd, L.ln_ffn.g, L.ln_ffn.b, hbuf);
        gemm(stream, hbuf, L.ff1.w, L.ff1.b, ffb, M, FF_, Dh, Dh, FF_, 0, 1, 0);
        gemm(stream, ffb, L.ff2.w, L.ff2.b, xd, M, Dh, FF_, FF_, Dh, 0, 0, 1);
    }

    // ---------------- tied-embedding logits: (512,768) @ E^T -> (512,32000)
    gemm(stream, xd, embed, nullptr, (float*)d_out, M, V_, Dh, Dh, V_, 1, 0, 0);
}
